// Rvt_58823872086215
// MI455X (gfx1250) — matrix operand order repaired and checked by reading, not compiled, not run on hardware
//
#include <hip/hip_runtime.h>
#include <hip/hip_bf16.h>
#include <stdint.h>

// ---------------------------------------------------------------------------
// CDNA5 (gfx1250) WMMA types & helpers
// ---------------------------------------------------------------------------
typedef __attribute__((ext_vector_type(16))) __bf16 v16bf;
typedef __attribute__((ext_vector_type(8)))  float  v8f;

union Frag16 {
    uint4          q[2];
    unsigned short us[16];
    v16bf          v;
};

__device__ __forceinline__ unsigned short f2bf(float x) {
    union { float f; unsigned u; } c; c.f = x;
    unsigned u = c.u + 0x7FFFu + ((c.u >> 16) & 1u);   // round-to-nearest-even
    return (unsigned short)(u >> 16);
}
__device__ __forceinline__ float bf2f(unsigned short b) {
    union { unsigned u; float f; } c; c.u = ((unsigned)b) << 16; return c.f;
}
__device__ __forceinline__ v8f v8f_zero() {
    v8f z = {0.f, 0.f, 0.f, 0.f, 0.f, 0.f, 0.f, 0.f}; return z;
}
__device__ __forceinline__ v8f wmma_bf16(v16bf a, v16bf b, v8f c) {
    // v_wmma_f32_16x16x32_bf16  (D = A x B + C), f32 accumulation
    return __builtin_amdgcn_wmma_f32_16x16x32_bf16(
        false, a, false, b, (short)0, c, false, false);
}

// CDNA5 async global->LDS copy (16 bytes, per-lane addresses), ASYNCcnt-tracked
__device__ __forceinline__ void async_copy_b128(unsigned lds_byte_off,
                                                const void* gptr) {
    asm volatile("global_load_async_to_lds_b128 %0, %1, off"
                 :: "v"(lds_byte_off),
                    "v"((unsigned long long)(uintptr_t)gptr)
                 : "memory");
}
__device__ __forceinline__ void wait_async0() {
    asm volatile("s_wait_asynccnt 0x0" ::: "memory");
}
__device__ __forceinline__ unsigned lds_off(const void* p) {
    return (unsigned)(uintptr_t)p;   // low 32 bits of generic ptr = LDS offset
}

// A-operand (16x32 bf16) from LDS, row-major [rows][stride], per ISA layout:
//  lane<16 : M=lane,    K chunks {0..7},{16..23};  lane>=16: M=lane-16, K {8..15},{24..31}
__device__ __forceinline__ v16bf load_a_s(const unsigned short* lds, int mBase,
                                          int k0, int stride) {
    int lane = threadIdx.x & 31;
    int hi   = lane >> 4;
    const uint4* p = (const uint4*)(lds + (mBase + (lane & 15)) * stride + k0);
    Frag16 f;
    f.q[0] = p[hi];
    f.q[1] = p[hi + 2];
    return f.v;
}
// B-operand (32x16 bf16) from LDS stored N-major [n][stride]: lane<16 reads
// K chunks {0..7},{16..23}; lane>=16 reads K {8..15},{24..31}
__device__ __forceinline__ v16bf load_b_s(const unsigned short* lds, int nBase,
                                          int k0, int stride) {
    int lane = threadIdx.x & 31;
    int hi   = lane >> 4;
    const uint4* p = (const uint4*)(lds + (nBase + (lane & 15)) * stride + k0);
    Frag16 f;
    f.q[0] = p[hi];
    f.q[1] = p[hi + 2];
    return f.v;
}

// ---------------------------------------------------------------------------
// Small prep kernels
// ---------------------------------------------------------------------------
__global__ void cvt_bf16_k(const float* __restrict__ in,
                           unsigned short* __restrict__ out, int n) {
    int i = blockIdx.x * 256 + threadIdx.x;
    if (i < n) out[i] = f2bf(in[i]);
}

// w[o][c][t] (t=0..8) -> wpack[o][t*256 + c], bf16.  O=C=256.
__global__ void pack3x3_k(const float* __restrict__ w,
                          unsigned short* __restrict__ out) {
    int idx = blockIdx.x * 256 + threadIdx.x;          // 256*2304 total
    if (idx >= 256 * 2304) return;
    int o = idx / 2304, k = idx % 2304;
    int t = k >> 8, c = k & 255;
    out[idx] = f2bf(w[(o * 256 + c) * 9 + t]);
}

// h[n][c][q] = h0[c]  (broadcast), bf16
__global__ void init_h_k(const float* __restrict__ h0,
                         unsigned short* __restrict__ h) {
    int idx = blockIdx.x * 256 + threadIdx.x;          // 8*256*1600
    if (idx >= 8 * 256 * 1600) return;
    int c = (idx / 1600) & 255;
    h[idx] = f2bf(h0[c]);
}

// ---------------------------------------------------------------------------
// Unified WMMA GEMM:  out[n][m][p] = sum_k A[m][k] * B_n[k][p]  (+bias)
//   M = 256 always.  Block tile 64x64, 128 threads = 4 wave32, 16 rows/wave.
//   A tile streamed with global_load_async_to_lds_b128.
//   ldo = output row stride (>= Np); pad columns [Np, ldo) are zero-filled.
// ---------------------------------------------------------------------------
enum { B_F32 = 0, B_CONV_SAME = 1, B_CONV_VALID = 2, B_CONCAT = 3 };

template <int BMODE, bool OUTF32>
__global__ __launch_bounds__(128) void gemm_k(
    const unsigned short* __restrict__ Apack, int Ktot,
    const void* __restrict__ Bsrc0, long bStride0,
    const unsigned short* __restrict__ Bsrc1, long bStride1,
    const float* __restrict__ bias,
    void* __restrict__ outp, int Np, int ldo) {
    __shared__ __align__(16) unsigned short ldsA[64 * 32];  // [m][k]
    __shared__ __align__(16) unsigned short ldsB[64 * 32];  // [p][k] (N-major)

    int tid  = threadIdx.x;
    int wave = tid >> 5, lane = tid & 31;
    int n  = blockIdx.z;
    int m0 = blockIdx.y * 64;
    int p0 = blockIdx.x * 64;

    v8f acc[4];
    for (int j = 0; j < 4; ++j) acc[j] = v8f_zero();

    unsigned ldsAoff = lds_off(ldsA) + tid * 32;   // 32B per thread
    int Kt = Ktot >> 5;
    for (int kt = 0; kt < Kt; ++kt) {
        __syncthreads();
        // ---- A tile: 64x32 bf16, async global->LDS (2 x B128 per thread)
        {
            const unsigned short* s = Apack + (long)(m0 + (tid >> 1)) * Ktot +
                                      kt * 32 + (tid & 1) * 16;
            async_copy_b128(ldsAoff, s);
            async_copy_b128(ldsAoff + 16, (const unsigned char*)s + 16);
        }
        // ---- B tile: [p][k] N-major (gather / implicit im2col)
        {
            int kk = tid >> 2, pseg = tid & 3;
            for (int i = 0; i < 16; ++i) {
                int p = pseg * 16 + i;
                int pp = p0 + p;
                unsigned short val = 0;
                if (BMODE == B_F32) {
                    const float* s = (const float*)Bsrc0 + (long)n * bStride0;
                    if (pp < Np) val = f2bf(s[(long)(kt * 32 + kk) * Np + pp]);
                } else if (BMODE == B_CONV_SAME) {
                    const unsigned short* s =
                        (const unsigned short*)Bsrc0 + (long)n * bStride0;
                    int t = kt >> 3, c = ((kt & 7) << 5) + kk;
                    int oh = pp / 40, ow = pp - oh * 40;
                    int sh = oh + t / 3 - 1, sw = ow + t % 3 - 1;
                    if (sh >= 0 && sh < 40 && sw >= 0 && sw < 40)
                        val = s[(long)c * 1600 + sh * 40 + sw];
                } else if (BMODE == B_CONV_VALID) {
                    const unsigned short* s =
                        (const unsigned short*)Bsrc0 + (long)n * bStride0;
                    if (pp < Np) {
                        int t = kt >> 3, c = ((kt & 7) << 5) + kk;
                        int oh = pp / 38, ow = pp - oh * 38;
                        int sh = oh + t / 3, sw = ow + t % 3;
                        val = s[(long)c * 1600 + sh * 40 + sw];
                    }
                } else {  // B_CONCAT: k<256 -> xp, else abuf
                    int kg = kt * 32 + kk;
                    if (kg < 256) {
                        const unsigned short* s =
                            (const unsigned short*)Bsrc0 + (long)n * bStride0;
                        val = s[(long)kg * 1600 + pp];
                    } else {
                        const unsigned short* s = Bsrc1 + (long)n * bStride1;
                        val = s[(long)(kg - 256) * 1600 + pp];
                    }
                }
                ldsB[p * 32 + kk] = val;
            }
        }
        wait_async0();
        __syncthreads();
        v16bf a = load_a_s(ldsA, wave * 16, 0, 32);
#pragma unroll
        for (int j = 0; j < 4; ++j) {
            v16bf b = load_b_s(ldsB, j * 16, 0, 32);
            acc[j]  = wmma_bf16(a, b, acc[j]);
        }
    }
    // ---- epilogue (C layout: n = lane&15, m = vgpr + 8*(lane>>4))
    int hi = lane >> 4, nn = lane & 15;
    for (int j = 0; j < 4; ++j) {
        int p = p0 + j * 16 + nn;
        if (p >= ldo) continue;
#pragma unroll
        for (int v = 0; v < 8; ++v) {
            int m   = m0 + wave * 16 + hi * 8 + v;
            float r = (p < Np) ? (acc[j][v] + (bias ? bias[m] : 0.f)) : 0.f;
            long idx = (long)n * 256 * ldo + (long)m * ldo + p;
            if (OUTF32) ((float*)outp)[idx] = r;
            else        ((unsigned short*)outp)[idx] = f2bf(r);
        }
    }
}

// ---------------------------------------------------------------------------
// Streaming attention (single pass, no max-subtraction: logits are O(1)):
//   S[q,d] = sum_c Q[c,q] K[c,d];  P = exp(S);  acc[q,c] += P * V^T;
//   rs[q]  += rowsum(P);  a[c,q] = acc[q,c] / rs[q]
// Block: 128 thr = 4 waves; 64 q rows/block (16/wave); d tiles of 32; C=256.
// K/V rows have padded stride ldk (zero-filled), so V tiles stream via
// async B128 copies and K tiles need no bounds checks.
// ---------------------------------------------------------------------------
__global__ __launch_bounds__(128) void attn_k(
    const unsigned short* __restrict__ q,     // [n][256][1600]
    const unsigned short* __restrict__ kmat,  // [n][256][ldk]
    const unsigned short* __restrict__ vmat,  // [n][256][ldk]
    unsigned short* __restrict__ abuf,        // [n][512][1600]
    int D, int ldk, int chanOff) {
    __shared__ __align__(16) unsigned short ldsK[32 * 256];  // [d][c] N-major (S)
    __shared__ __align__(16) unsigned short ldsV[256 * 32];  // [c][d] N-major (PV)
    __shared__ __align__(16) unsigned short ldsP[4 * 16 * 32];  // per-wave P

    int tid = threadIdx.x, wave = tid >> 5, lane = tid & 31;
    int hi = lane >> 4, nn = lane & 15;
    int n  = blockIdx.y;
    int q0 = blockIdx.x * 64;

    const unsigned short* qn = q    + (long)n * 256 * 1600;
    const unsigned short* kn = kmat + (long)n * 256 * ldk;
    const unsigned short* vn = vmat + (long)n * 256 * ldk;

    // Preload Q A-operand fragments straight from global (ISA A-layout gather)
    int qrow = q0 + wave * 16 + nn;
    v16bf qa[8];
#pragma unroll
    for (int c0 = 0; c0 < 8; ++c0) {
        Frag16 f;
#pragma unroll
        for (int j = 0; j < 8; ++j) {
            int c = c0 * 32 + 2 * j + ((j >= 4) ? 8 : 0) + hi * 8;
            f.us[2 * j]     = qn[(long)c * 1600 + qrow];
            f.us[2 * j + 1] = qn[(long)(c + 1) * 1600 + qrow];
        }
        qa[c0] = f.v;
    }

    v8f acc[16];
    for (int cf = 0; cf < 16; ++cf) acc[cf] = v8f_zero();
    float rs[8] = {0.f, 0.f, 0.f, 0.f, 0.f, 0.f, 0.f, 0.f};

    int c0r = tid * 2;                       // this thread's 2 channel rows
    unsigned vdst0 = lds_off(ldsV) + c0r * 64;
    int Dt = (D + 31) >> 5;
    for (int dt = 0; dt < Dt; ++dt) {
        int d0 = dt * 32;
        __syncthreads();
        // ---- V tile: async global->LDS, [c][32d], 4 x B128 per row
        for (int r = 0; r < 2; ++r) {
            const unsigned short* vrow = vn + (long)(c0r + r) * ldk + d0;
            unsigned vd = vdst0 + r * 64;
#pragma unroll
            for (int j = 0; j < 4; ++j)
                async_copy_b128(vd + j * 16, (const unsigned char*)vrow + j * 16);
        }
        // ---- K tile: transposed scatter [d][c] (rows are zero-padded)
        for (int r = 0; r < 2; ++r) {
            int c = c0r + r;
            const unsigned short* krow = kn + (long)c * ldk + d0;
            for (int dd = 0; dd < 32; ++dd) ldsK[dd * 256 + c] = krow[dd];
            if (d0 + 32 < D) {  // hint next tile (global_prefetch_b8)
                __builtin_prefetch(krow + 32, 0, 1);
                __builtin_prefetch(vn + (long)c * ldk + d0 + 32, 0, 1);
            }
        }
        wait_async0();
        __syncthreads();
        // S = Q * K^T for this wave's 16 q rows, 32 d cols; P = exp(S)
#pragma unroll
        for (int nf = 0; nf < 2; ++nf) {
            v8f s = v8f_zero();
#pragma unroll
            for (int c0 = 0; c0 < 8; ++c0)
                s = wmma_bf16(qa[c0], load_b_s(ldsK, nf * 16, c0 * 32, 256), s);
            int dcol = d0 + nf * 16 + nn;
#pragma unroll
            for (int v = 0; v < 8; ++v) {
                float pv = (dcol < D) ? __expf(s[v]) : 0.f;
                rs[v] += pv;
                ldsP[wave * 512 + (hi * 8 + v) * 32 + nf * 16 + nn] = f2bf(pv);
            }
        }
        // acc += P * V^T  (16 c-fragments, K=32)
        v16bf pa = load_a_s(ldsP + wave * 512, 0, 0, 32);
#pragma unroll
        for (int cf = 0; cf < 16; ++cf) {
            v16bf b = load_b_s(ldsV, cf * 16, 0, 32);
            acc[cf] = wmma_bf16(pa, b, acc[cf]);
        }
    }
    // Row-sum reduction over the 16 n-lanes of each half-wave
#pragma unroll
    for (int v = 0; v < 8; ++v) {
        float r = rs[v];
        r += __shfl_xor(r, 1, 32);
        r += __shfl_xor(r, 2, 32);
        r += __shfl_xor(r, 4, 32);
        r += __shfl_xor(r, 8, 32);
        rs[v] = 1.f / r;
    }
    // Store a[c][q] (transposed scatter), bf16
    unsigned short* an = abuf + (long)n * 512 * 1600;
    for (int cf = 0; cf < 16; ++cf) {
        int c = chanOff + cf * 16 + nn;
#pragma unroll
        for (int v = 0; v < 8; ++v) {
            int qq = q0 + wave * 16 + hi * 8 + v;
            an[(long)c * 1600 + qq] = f2bf(acc[cf][v] * rs[v]);
        }
    }
}

// ---------------------------------------------------------------------------
// Fused GRU-style gate: z = sigmoid(Wz*a_h + bz), hc = tanh(Wh*a_h + bh),
//                       h = z*h + (1-z)*hc       (dual-GEMM, shared B tile)
// ---------------------------------------------------------------------------
__global__ __launch_bounds__(128) void gate_k(
    const unsigned short* __restrict__ wz, const unsigned short* __restrict__ wh,
    const float* __restrict__ bz, const float* __restrict__ bh,
    const unsigned short* __restrict__ abuf,  // [n][512][1600]
    unsigned short* __restrict__ h) {         // [n][256][1600] in/out
    __shared__ __align__(16) unsigned short ldsAz[64 * 32];
    __shared__ __align__(16) unsigned short ldsAh[64 * 32];
    __shared__ __align__(16) unsigned short ldsB[64 * 32];

    int tid = threadIdx.x, wave = tid >> 5, lane = tid & 31;
    int n = blockIdx.z, m0 = blockIdx.y * 64, p0 = blockIdx.x * 64;
    const unsigned short* ab = abuf + (long)n * 512 * 1600;

    v8f az[4], ah[4];
    for (int j = 0; j < 4; ++j) { az[j] = v8f_zero(); ah[j] = v8f_zero(); }

    unsigned oz = lds_off(ldsAz) + tid * 32;
    unsigned oh2 = lds_off(ldsAh) + tid * 32;
    for (int kt = 0; kt < 16; ++kt) {
        __syncthreads();
        {
            long src = (long)(m0 + (tid >> 1)) * 512 + kt * 32 + (tid & 1) * 16;
            async_copy_b128(oz,       wz + src);
            async_copy_b128(oz + 16,  (const unsigned char*)(wz + src) + 16);
            async_copy_b128(oh2,      wh + src);
            async_copy_b128(oh2 + 16, (const unsigned char*)(wh + src) + 16);
        }
        {
            int kk = tid >> 2, pseg = tid & 3;
            for (int i = 0; i < 16; ++i) {
                int p = pseg * 16 + i;
                ldsB[p * 32 + kk] = ab[(long)(kt * 32 + kk) * 1600 + p0 + p];
            }
        }
        wait_async0();
        __syncthreads();
        v16bf a1 = load_a_s(ldsAz, wave * 16, 0, 32);
        v16bf a2 = load_a_s(ldsAh, wave * 16, 0, 32);
#pragma unroll
        for (int j = 0; j < 4; ++j) {
            v16bf b = load_b_s(ldsB, j * 16, 0, 32);
            az[j] = wmma_bf16(a1, b, az[j]);
            ah[j] = wmma_bf16(a2, b, ah[j]);
        }
    }
    int hi = lane >> 4, nn = lane & 15;
    unsigned short* hn = h + (long)n * 256 * 1600;
    for (int j = 0; j < 4; ++j) {
        int p = p0 + j * 16 + nn;
#pragma unroll
        for (int v = 0; v < 8; ++v) {
            int m = m0 + wave * 16 + hi * 8 + v;
            float zv = 1.f / (1.f + __expf(-(az[j][v] + bz[m])));
            float hc = tanhf(ah[j][v] + bh[m]);
            long idx = (long)m * 1600 + p;
            float ho = bf2f(hn[idx]);
            hn[idx] = f2bf(zv * ho + (1.f - zv) * hc);
        }
    }
}

// ---------------------------------------------------------------------------
// Host orchestration
// ---------------------------------------------------------------------------
extern "C" void kernel_launch(void* const* d_in, const int* in_sizes, int n_in,
                              void* d_out, int out_size, void* d_ws,
                              size_t ws_size, hipStream_t stream) {
    (void)in_sizes; (void)n_in; (void)out_size; (void)ws_size;
    const float* x     = (const float*)d_in[0];
    const float* h0    = (const float*)d_in[1];
    const float* w_px  = (const float*)d_in[2];
    const float* b_px  = (const float*)d_in[3];
    const float* w_qx  = (const float*)d_in[4];
    const float* w_qh  = (const float*)d_in[5];
    const float* w_kx  = (const float*)d_in[6];
    const float* w_kh  = (const float*)d_in[7];
    const float* w_vx  = (const float*)d_in[8];
    const float* w_vh  = (const float*)d_in[9];
    const float* w_z   = (const float*)d_in[10];
    const float* b_z   = (const float*)d_in[11];
    const float* w_h   = (const float*)d_in[12];
    const float* b_h   = (const float*)d_in[13];
    const float* w_out = (const float*)d_in[14];
    const float* b_out = (const float*)d_in[15];

    unsigned char* ws = (unsigned char*)d_ws;
    size_t off = 0;
    auto alloc = [&](size_t bytes) -> void* {
        void* p = ws + off;
        off += (bytes + 255) & ~(size_t)255;
        return p;
    };
    const int NQ = 1600, ND = 1444, NDP = 1472, NB = 8;  // NDP: padded k/v stride
    unsigned short* wpx_p  = (unsigned short*)alloc(256 * 128  * 2);
    unsigned short* wqx_p  = (unsigned short*)alloc(256 * 2304 * 2);
    unsigned short* wqh_p  = (unsigned short*)alloc(256 * 2304 * 2);
    unsigned short* wkx_p  = (unsigned short*)alloc(256 * 2304 * 2);
    unsigned short* wkh_p  = (unsigned short*)alloc(256 * 2304 * 2);
    unsigned short* wvx_p  = (unsigned short*)alloc(256 * 2304 * 2);
    unsigned short* wvh_p  = (unsigned short*)alloc(256 * 2304 * 2);
    unsigned short* wz_p   = (unsigned short*)alloc(256 * 512  * 2);
    unsigned short* wh_p   = (unsigned short*)alloc(256 * 512  * 2);
    unsigned short* wout_p = (unsigned short*)alloc(256 * 768  * 2);
    unsigned short* xp   = (unsigned short*)alloc((size_t)NB * 256 * NQ  * 2);
    unsigned short* qx   = (unsigned short*)alloc((size_t)NB * 256 * NQ  * 2);
    unsigned short* kx   = (unsigned short*)alloc((size_t)NB * 256 * NDP * 2);
    unsigned short* vx   = (unsigned short*)alloc((size_t)NB * 256 * NDP * 2);
    unsigned short* qh   = (unsigned short*)alloc((size_t)NB * 256 * NQ  * 2);
    unsigned short* kh   = (unsigned short*)alloc((size_t)NB * 256 * NDP * 2);
    unsigned short* vh   = (unsigned short*)alloc((size_t)NB * 256 * NDP * 2);
    unsigned short* hbuf = (unsigned short*)alloc((size_t)NB * 256 * NQ  * 2);
    unsigned short* ab   = (unsigned short*)alloc((size_t)NB * 512 * NQ  * 2);

    // ---- weight packing / conversion
    cvt_bf16_k<<<(256 * 128 + 255) / 256, 256, 0, stream>>>(w_px, wpx_p, 256 * 128);
    pack3x3_k<<<2304, 256, 0, stream>>>(w_qx, wqx_p);
    pack3x3_k<<<2304, 256, 0, stream>>>(w_qh, wqh_p);
    pack3x3_k<<<2304, 256, 0, stream>>>(w_kx, wkx_p);
    pack3x3_k<<<2304, 256, 0, stream>>>(w_kh, wkh_p);
    pack3x3_k<<<2304, 256, 0, stream>>>(w_vx, wvx_p);
    pack3x3_k<<<2304, 256, 0, stream>>>(w_vh, wvh_p);
    cvt_bf16_k<<<(256 * 512 + 255) / 256, 256, 0, stream>>>(w_z, wz_p, 256 * 512);
    cvt_bf16_k<<<(256 * 512 + 255) / 256, 256, 0, stream>>>(w_h, wh_p, 256 * 512);
    cvt_bf16_k<<<(256 * 768 + 255) / 256, 256, 0, stream>>>(w_out, wout_p, 256 * 768);
    init_h_k<<<(NB * 256 * NQ + 255) / 256, 256, 0, stream>>>(h0, hbuf);

    dim3 gQ(25, 4, NB), gD(23, 4, NB), blk(128);

    // xp = w_px * x + b_px   (K=128, f32 source)
    gemm_k<B_F32, false><<<gQ, blk, 0, stream>>>(
        wpx_p, 128, x, (long)128 * NQ, nullptr, 0, b_px, xp, NQ, NQ);

    // step-invariant x-side convs
    gemm_k<B_CONV_SAME,  false><<<gQ, blk, 0, stream>>>(
        wqx_p, 2304, xp, (long)256 * NQ, nullptr, 0, nullptr, qx, NQ, NQ);
    gemm_k<B_CONV_VALID, false><<<gD, blk, 0, stream>>>(
        wkx_p, 2304, xp, (long)256 * NQ, nullptr, 0, nullptr, kx, ND, NDP);
    gemm_k<B_CONV_VALID, false><<<gD, blk, 0, stream>>>(
        wvx_p, 2304, xp, (long)256 * NQ, nullptr, 0, nullptr, vx, ND, NDP);

    // ---- step 1: only the q_h attention side feeds the gate
    gemm_k<B_CONV_SAME,  false><<<gQ, blk, 0, stream>>>(
        wqh_p, 2304, hbuf, (long)256 * NQ, nullptr, 0, nullptr, qh, NQ, NQ);
    gemm_k<B_CONV_VALID, false><<<gD, blk, 0, stream>>>(
        wkh_p, 2304, hbuf, (long)256 * NQ, nullptr, 0, nullptr, kh, ND, NDP);
    gemm_k<B_CONV_VALID, false><<<gD, blk, 0, stream>>>(
        wvh_p, 2304, hbuf, (long)256 * NQ, nullptr, 0, nullptr, vh, ND, NDP);
    attn_k<<<dim3(25, NB), blk, 0, stream>>>(qh, kx, vx, ab, ND, NDP, 0);
    attn_k<<<dim3(25, NB), blk, 0, stream>>>(qh, kh, vh, ab, ND, NDP, 256);
    gate_k<<<gQ, blk, 0, stream>>>(wz_p, wh_p, b_z, b_h, ab, hbuf);

    // ---- step 2: only the q_x attention side feeds the output
    gemm_k<B_CONV_VALID, false><<<gD, blk, 0, stream>>>(
        wkh_p, 2304, hbuf, (long)256 * NQ, nullptr, 0, nullptr, kh, ND, NDP);
    gemm_k<B_CONV_VALID, false><<<gD, blk, 0, stream>>>(
        wvh_p, 2304, hbuf, (long)256 * NQ, nullptr, 0, nullptr, vh, ND, NDP);
    attn_k<<<dim3(25, NB), blk, 0, stream>>>(qx, kx, vx, ab, ND, NDP, 0);
    attn_k<<<dim3(25, NB), blk, 0, stream>>>(qx, kh, vh, ab, ND, NDP, 256);

    // out = w_out * concat(xp, a_x) + b_out   (K=768, f32 output)
    gemm_k<B_CONCAT, true><<<gQ, blk, 0, stream>>>(
        wout_p, 768, xp, (long)256 * NQ, ab, (long)512 * NQ, b_out, d_out,
        NQ, NQ);
}
